// FullRankRNN_56779467653296
// MI455X (gfx1250) — matrix operand order repaired and checked by reading, not compiled, not run on hardware
//
#include <hip/hip_runtime.h>
#include <math.h>

typedef __attribute__((ext_vector_type(2)))  float  v2f;
typedef __attribute__((ext_vector_type(8)))  float  v8f;
typedef __attribute__((ext_vector_type(16))) __bf16 v16bf;

#define HIDDEN 512
#define INSZ   256
#define TSTEPS 1024
#define BATCH  64
#define ALPHA  0.1f
#define HPAD   516   // fp32 h rows: +4 floats -> 4-bank skew
#define ABST   520   // bf16 a rows: +8 bf16  -> 4-bank skew, 16B aligned rows

union BF16x16 {
  v16bf v;
  uint4 u[2];
};

// ---------------------------------------------------------------------------
// Kernel 1: xu[m, n] = sum_k x[m, k] * U[n, k]  ->  written into d_out.
// Memory-bound (~192 MB of HBM traffic); fp32 WMMA 16x16x4, one tile/wave.
// ---------------------------------------------------------------------------
__global__ void __launch_bounds__(256)
xu_gemm_kernel(const float* __restrict__ x,
               const float* __restrict__ U,
               float* __restrict__ out) {
  const int lane  = threadIdx.x & 31;
  const int wave  = threadIdx.x >> 5;
  const int m0    = blockIdx.x << 4;
  const int n0    = ((blockIdx.y << 3) + wave) << 4;
  const int r     = lane & 15;
  const int khalf = (lane >> 4) << 1;

  const float* __restrict__ xrow = x + (size_t)(m0 + r) * INSZ + khalf;
  const float* __restrict__ urow = U + (size_t)(n0 + r) * INSZ + khalf;

  v8f acc = {};
#pragma unroll 8
  for (int k0 = 0; k0 < INSZ; k0 += 4) {
    v2f af = *(const v2f*)(xrow + k0);
    v2f bf = *(const v2f*)(urow + k0);
    acc = __builtin_amdgcn_wmma_f32_16x16x4_f32(
        false, af, false, bf, (short)0, acc, false, false);
  }

  const int mbase = m0 + ((lane >> 4) << 3);
  float* __restrict__ orow = out + (size_t)mbase * HIDDEN + n0 + r;
#pragma unroll
  for (int j = 0; j < 8; ++j) {
    orow[(size_t)j * HIDDEN] = acc[j];
  }
}

// ---------------------------------------------------------------------------
// Kernel 2: split W (fp32) -> W_hi + W_lo (bf16), hi/lo residual decomposition.
// a@W ~= a_hi@W_hi + a_lo@W_hi + a_hi@W_lo keeps ~2^-16 relative accuracy.
// ---------------------------------------------------------------------------
__global__ void __launch_bounds__(256)
wsplit_kernel(const float* __restrict__ W,
              __bf16* __restrict__ Wh,
              __bf16* __restrict__ Wl) {
  const int i = blockIdx.x * 256 + threadIdx.x;   // HIDDEN*HIDDEN total
  const float w = W[i];
  const __bf16 hi = (__bf16)w;
  Wh[i] = hi;
  Wl[i] = (__bf16)(w - (float)hi);
}

// ---------------------------------------------------------------------------
// Kernel 3 (fast path): recurrence with V_WMMA_F32_16X16X32_BF16.
// 4 blocks x 256 threads; block owns a 16-batch tile; wave owns 64 N-columns
// (4 tiles). Per k-tile of 32: 2 LDS b128 loads per A fragment (hi and lo),
// one contiguous 32B W-row read per B fragment, 3 bf16 WMMAs per n-tile.
// ---------------------------------------------------------------------------
__global__ void __launch_bounds__(256)
rnn_recur_bf16_kernel(const __bf16* __restrict__ Wh,
                      const __bf16* __restrict__ Wl,
                      float* __restrict__ out) {
  extern __shared__ char smemraw[];
  float*  __restrict__ hbuf = (float*)smemraw;                         // [16][HPAD]
  __bf16* __restrict__ ahi  = (__bf16*)(smemraw + 16 * HPAD * 4);      // [16][ABST]
  __bf16* __restrict__ alo  = ahi + 16 * ABST;                          // [16][ABST]

  const int tid   = threadIdx.x;
  const int lane  = tid & 31;
  const int wave  = tid >> 5;
  const int b0    = blockIdx.x << 4;
  const int r     = lane & 15;
  const int subA  = (lane >> 4) << 4;   // A: upper lane-half -> K+16
  const int subB  = (lane >> 4) << 4;   // B: upper lane-half -> K+16
  const int nbase = wave << 6;

  for (int i = tid; i < 16 * HPAD; i += 256) hbuf[i] = 0.0f;
  __syncthreads();

  const __bf16* __restrict__ ahp = ahi + r * ABST + subA;
  const __bf16* __restrict__ alp = alo + r * ABST + subA;
  const __bf16* __restrict__ wh0 = Wh + (size_t)(nbase +  0 + r) * HIDDEN + subB;
  const __bf16* __restrict__ wh1 = Wh + (size_t)(nbase + 16 + r) * HIDDEN + subB;
  const __bf16* __restrict__ wh2 = Wh + (size_t)(nbase + 32 + r) * HIDDEN + subB;
  const __bf16* __restrict__ wh3 = Wh + (size_t)(nbase + 48 + r) * HIDDEN + subB;
  const __bf16* __restrict__ wl0 = Wl + (size_t)(nbase +  0 + r) * HIDDEN + subB;
  const __bf16* __restrict__ wl1 = Wl + (size_t)(nbase + 16 + r) * HIDDEN + subB;
  const __bf16* __restrict__ wl2 = Wl + (size_t)(nbase + 32 + r) * HIDDEN + subB;
  const __bf16* __restrict__ wl3 = Wl + (size_t)(nbase + 48 + r) * HIDDEN + subB;

  for (int t = 0; t < TSTEPS; ++t) {
    // a = tanh(h) split into bf16 hi + lo residual
    for (int i = tid; i < 16 * HIDDEN; i += 256) {
      const int row = i >> 9;
      const int col = i & 511;
      const float a  = tanhf(hbuf[row * HPAD + col]);
      const __bf16 h = (__bf16)a;
      ahi[row * ABST + col] = h;
      alo[row * ABST + col] = (__bf16)(a - (float)h);
    }
    __syncthreads();

    v8f acc0 = {}, acc1 = {}, acc2 = {}, acc3 = {};
#pragma unroll 4
    for (int k0 = 0; k0 < HIDDEN; k0 += 32) {
      BF16x16 AH, AL;
      AH.u[0] = *(const uint4*)(ahp + k0);
      AH.u[1] = *(const uint4*)(ahp + k0 + 8);
      AL.u[0] = *(const uint4*)(alp + k0);
      AL.u[1] = *(const uint4*)(alp + k0 + 8);

      BF16x16 BH, BL;
      BH.u[0] = *(const uint4*)(wh0 + k0); BH.u[1] = *(const uint4*)(wh0 + k0 + 8);
      BL.u[0] = *(const uint4*)(wl0 + k0); BL.u[1] = *(const uint4*)(wl0 + k0 + 8);
      acc0 = __builtin_amdgcn_wmma_f32_16x16x32_bf16(false, AH.v, false, BH.v, (short)0, acc0, false, false);
      acc0 = __builtin_amdgcn_wmma_f32_16x16x32_bf16(false, AL.v, false, BH.v, (short)0, acc0, false, false);
      acc0 = __builtin_amdgcn_wmma_f32_16x16x32_bf16(false, AH.v, false, BL.v, (short)0, acc0, false, false);

      BH.u[0] = *(const uint4*)(wh1 + k0); BH.u[1] = *(const uint4*)(wh1 + k0 + 8);
      BL.u[0] = *(const uint4*)(wl1 + k0); BL.u[1] = *(const uint4*)(wl1 + k0 + 8);
      acc1 = __builtin_amdgcn_wmma_f32_16x16x32_bf16(false, AH.v, false, BH.v, (short)0, acc1, false, false);
      acc1 = __builtin_amdgcn_wmma_f32_16x16x32_bf16(false, AL.v, false, BH.v, (short)0, acc1, false, false);
      acc1 = __builtin_amdgcn_wmma_f32_16x16x32_bf16(false, AH.v, false, BL.v, (short)0, acc1, false, false);

      BH.u[0] = *(const uint4*)(wh2 + k0); BH.u[1] = *(const uint4*)(wh2 + k0 + 8);
      BL.u[0] = *(const uint4*)(wl2 + k0); BL.u[1] = *(const uint4*)(wl2 + k0 + 8);
      acc2 = __builtin_amdgcn_wmma_f32_16x16x32_bf16(false, AH.v, false, BH.v, (short)0, acc2, false, false);
      acc2 = __builtin_amdgcn_wmma_f32_16x16x32_bf16(false, AL.v, false, BH.v, (short)0, acc2, false, false);
      acc2 = __builtin_amdgcn_wmma_f32_16x16x32_bf16(false, AH.v, false, BL.v, (short)0, acc2, false, false);

      BH.u[0] = *(const uint4*)(wh3 + k0); BH.u[1] = *(const uint4*)(wh3 + k0 + 8);
      BL.u[0] = *(const uint4*)(wl3 + k0); BL.u[1] = *(const uint4*)(wl3 + k0 + 8);
      acc3 = __builtin_amdgcn_wmma_f32_16x16x32_bf16(false, AH.v, false, BH.v, (short)0, acc3, false, false);
      acc3 = __builtin_amdgcn_wmma_f32_16x16x32_bf16(false, AL.v, false, BH.v, (short)0, acc3, false, false);
      acc3 = __builtin_amdgcn_wmma_f32_16x16x32_bf16(false, AH.v, false, BL.v, (short)0, acc3, false, false);
    }

    // h update + in-place output write (reads xu_t, writes h_t)
    const int mrow = (lane >> 4) << 3;
#pragma unroll
    for (int nt = 0; nt < 4; ++nt) {
      const v8f acc = (nt == 0) ? acc0 : (nt == 1) ? acc1 : (nt == 2) ? acc2 : acc3;
      const int n = nbase + (nt << 4) + r;
#pragma unroll
      for (int j = 0; j < 8; ++j) {
        const int row = mrow + j;
        const size_t gidx = ((size_t)(b0 + row) * TSTEPS + t) * HIDDEN + n;
        const float xu    = out[gidx];
        const float hprev = hbuf[row * HPAD + n];
        const float hn    = (1.0f - ALPHA) * hprev + ALPHA * (acc[j] + xu);
        out[gidx]            = hn;
        hbuf[row * HPAD + n] = hn;
      }
    }
    __syncthreads();
  }
}

// ---------------------------------------------------------------------------
// Fallback recurrence (fp32 WMMA 16x16x4), used only if d_ws can't hold the
// 1 MB W_hi/W_lo split. Same structure as round 1.
// ---------------------------------------------------------------------------
__global__ void __launch_bounds__(256)
rnn_recur_f32_kernel(const float* __restrict__ W,
                     float* __restrict__ out) {
  extern __shared__ char smemraw[];
  float* __restrict__ hbuf = (float*)smemraw;
  float* __restrict__ abuf = hbuf + 16 * HPAD;

  const int tid   = threadIdx.x;
  const int lane  = tid & 31;
  const int wave  = tid >> 5;
  const int b0    = blockIdx.x << 4;
  const int r     = lane & 15;
  const int khalf = (lane >> 4) << 1;
  const int nbase = wave << 6;

  for (int i = tid; i < 16 * HPAD; i += 256) hbuf[i] = 0.0f;
  __syncthreads();

  const float* __restrict__ a_lane = abuf + r * HPAD + khalf;
  const float* __restrict__ wl0 = W + (size_t)(nbase +  0 + r) * HIDDEN + khalf;
  const float* __restrict__ wl1 = W + (size_t)(nbase + 16 + r) * HIDDEN + khalf;
  const float* __restrict__ wl2 = W + (size_t)(nbase + 32 + r) * HIDDEN + khalf;
  const float* __restrict__ wl3 = W + (size_t)(nbase + 48 + r) * HIDDEN + khalf;

  for (int t = 0; t < TSTEPS; ++t) {
    for (int i = tid; i < 16 * HIDDEN; i += 256) {
      const int row = i >> 9;
      const int col = i & 511;
      abuf[row * HPAD + col] = tanhf(hbuf[row * HPAD + col]);
    }
    __syncthreads();

    v8f acc0 = {}, acc1 = {}, acc2 = {}, acc3 = {};
#pragma unroll 4
    for (int k0 = 0; k0 < HIDDEN; k0 += 4) {
      v2f af = *(const v2f*)(a_lane + k0);
      v2f b0f = *(const v2f*)(wl0 + k0);
      acc0 = __builtin_amdgcn_wmma_f32_16x16x4_f32(false, af, false, b0f, (short)0, acc0, false, false);
      v2f b1f = *(const v2f*)(wl1 + k0);
      acc1 = __builtin_amdgcn_wmma_f32_16x16x4_f32(false, af, false, b1f, (short)0, acc1, false, false);
      v2f b2f = *(const v2f*)(wl2 + k0);
      acc2 = __builtin_amdgcn_wmma_f32_16x16x4_f32(false, af, false, b2f, (short)0, acc2, false, false);
      v2f b3f = *(const v2f*)(wl3 + k0);
      acc3 = __builtin_amdgcn_wmma_f32_16x16x4_f32(false, af, false, b3f, (short)0, acc3, false, false);
    }

    const int mrow = (lane >> 4) << 3;
#pragma unroll
    for (int nt = 0; nt < 4; ++nt) {
      const v8f acc = (nt == 0) ? acc0 : (nt == 1) ? acc1 : (nt == 2) ? acc2 : acc3;
      const int n = nbase + (nt << 4) + r;
#pragma unroll
      for (int j = 0; j < 8; ++j) {
        const int row = mrow + j;
        const size_t gidx = ((size_t)(b0 + row) * TSTEPS + t) * HIDDEN + n;
        const float xu    = out[gidx];
        const float hprev = hbuf[row * HPAD + n];
        const float hn    = (1.0f - ALPHA) * hprev + ALPHA * (acc[j] + xu);
        out[gidx]            = hn;
        hbuf[row * HPAD + n] = hn;
      }
    }
    __syncthreads();
  }
}

// ---------------------------------------------------------------------------
extern "C" void kernel_launch(void* const* d_in, const int* in_sizes, int n_in,
                              void* d_out, int out_size, void* d_ws, size_t ws_size,
                              hipStream_t stream) {
  const float* x = (const float*)d_in[0];  // [64, 1024, 256]
  const float* W = (const float*)d_in[1];  // [512, 512]
  const float* U = (const float*)d_in[2];  // [512, 256]
  float* out = (float*)d_out;              // [64, 1024, 512]

  // Kernel 1: xu -> d_out
  dim3 gA(4096, 4);
  xu_gemm_kernel<<<gA, 256, 0, stream>>>(x, U, out);

  const size_t split_bytes = (size_t)2 * HIDDEN * HIDDEN * sizeof(__bf16); // 1 MB
  if (ws_size >= split_bytes) {
    __bf16* Wh = (__bf16*)d_ws;
    __bf16* Wl = Wh + (size_t)HIDDEN * HIDDEN;
    wsplit_kernel<<<dim3(HIDDEN * HIDDEN / 256), 256, 0, stream>>>(W, Wh, Wl);

    const size_t shmem = (size_t)16 * HPAD * 4 + (size_t)2 * 16 * ABST * 2;
    rnn_recur_bf16_kernel<<<dim3(4), 256, shmem, stream>>>(Wh, Wl, out);
  } else {
    const size_t shmem = (size_t)2 * 16 * HPAD * sizeof(float);
    rnn_recur_f32_kernel<<<dim3(4), 256, shmem, stream>>>(W, out);
  }
}
